// LaneATT_63763084477275
// MI455X (gfx1250) — matrix operand order repaired and checked by reading, not compiled, not run on hardware
//
#include <hip/hip_runtime.h>
#include <hip/hip_bf16.h>

typedef __attribute__((ext_vector_type(16))) _Float16 v16h;
typedef __attribute__((ext_vector_type(8)))  float    v8f;
typedef __attribute__((ext_vector_type(4)))  unsigned int v4u;

union FragH { v16h h; v4u q[2]; };

#define TILE_M 64
#define TILE_N 128
#define TILE_K 32
#define PI_F 3.14159265358979f

__constant__ float cAngles[17] = {15.f,22.f,30.f,39.f,49.f,60.f,72.f,80.f,90.f,
                                  100.f,108.f,120.f,131.f,141.f,150.f,158.f,165.f};

// ---------------- small utility kernels ----------------

__global__ void cvt_f32_f16_kernel(const float* __restrict__ src,
                                   _Float16* __restrict__ dst, int n) {
  int i = blockIdx.x * blockDim.x + threadIdx.x;
  if (i < n) dst[i] = (_Float16)src[i];
}

__global__ void bnfold_kernel(const float* __restrict__ g, const float* __restrict__ b,
                              const float* __restrict__ m, const float* __restrict__ v,
                              float* __restrict__ scale, float* __restrict__ shift, int c) {
  int i = blockIdx.x * blockDim.x + threadIdx.x;
  if (i < c) {
    float s = g[i] * rsqrtf(v[i] + 1e-5f);
    scale[i] = s;
    shift[i] = b[i] - m[i] * s;
  }
}

__global__ void maxpool_kernel(const _Float16* __restrict__ x, _Float16* __restrict__ out,
                               int B, int C, int IH, int IW, int OH, int OW) {
  int i = blockIdx.x * blockDim.x + threadIdx.x;
  int total = B * C * OH * OW;
  if (i >= total) return;
  int ow = i % OW; int t = i / OW; int oh = t % OH; t /= OH; int c = t % C; int b = t / C;
  float mx = -INFINITY;
  for (int r = 0; r < 3; ++r)
    for (int s = 0; s < 3; ++s) {
      int ih = oh * 2 - 1 + r, iw = ow * 2 - 1 + s;
      if ((unsigned)ih < (unsigned)IH && (unsigned)iw < (unsigned)IW) {
        float v = (float)x[(((size_t)b * C + c) * IH + ih) * IW + iw];
        mx = v > mx ? v : mx;
      }
    }
  out[i] = (_Float16)mx;
}

// ---------------- implicit-GEMM conv with WMMA f16 ----------------
// GEMM: M = out channels, N = B*OH*OW, K = C*R*S. Activations f16, accumulate f32.
// Block: 256 threads (8 wave32), block tile 64M x 128N, wave tile 16M x 64N (4 WMMA).
// Weight tile staged with global_load_async_to_lds_b128 when fully in-bounds.

template <int R, int S, int STRD, int PAD>
__global__ __launch_bounds__(256) void conv_wmma_kernel(
    const _Float16* __restrict__ x, const _Float16* __restrict__ w16,
    const float* __restrict__ scale, const float* __restrict__ shift,
    const _Float16* __restrict__ res, _Float16* __restrict__ out,
    int B, int C, int IH, int IW, int M, int OH, int OW, int relu) {
  __shared__ __align__(32) _Float16 aT[TILE_M * TILE_K];  // [m][k]
  __shared__ __align__(32) _Float16 bT[TILE_N * TILE_K];  // [n][k]
  const int tid  = threadIdx.x;
  const int lane = tid & 31;
  const int wave = tid >> 5;
  const int wm = wave >> 1;        // 0..3 (M sub-tile)
  const int wn = wave & 1;         // 0..1 (N sub-tile)
  const int mBase = blockIdx.y * TILE_M;
  const int nBase = blockIdx.x * TILE_N;
  constexpr int RS = R * S;
  const int K = C * RS;
  const int Ntot = B * OH * OW;
  const int l16 = lane & 15;
  const int ha = (lane >= 16) ? 1 : 0;   // A frag v4u offset
  const int hb = (lane >= 16) ? 2 : 0;   // B frag v4u offset
  const int hiM = (lane >= 16) ? 8 : 0;  // C/D row offset
  v8f acc[4] = {{}, {}, {}, {}};

  // ---- per-thread invariants: A staging (one m-row, 8 k's) ----
  const int mg = mBase + (tid >> 2);
  const int kOffA = (tid & 3) * 8;
  const _Float16* wrow = w16 + (size_t)mg * K;
  _Float16* aDst = &aT[(tid >> 2) * TILE_K + kOffA];
  const bool mOK = mg < M;
  const bool useAsync = ((K & 31) == 0) && (mBase + TILE_M <= M);

  // ---- per-thread invariants: B staging (one n-row, 16 k's) ----
  const int nRow = nBase + (tid >> 1);
  const int kOffB = (tid & 1) * 16;
  _Float16* bDst = &bT[(tid >> 1) * TILE_K + kOffB];
  const bool nOK = nRow < Ntot;
  int ow = 0, oh = 0, bb = 0;
  if (nOK) { ow = nRow % OW; int t2 = nRow / OW; oh = t2 % OH; bb = t2 / OH; }
  const int ihBase = oh * STRD - PAD;
  const int iwBase = ow * STRD - PAD;
  const _Float16* xb = x + (size_t)bb * C * IH * IW;

  const v4u* a4 = (const v4u*)aT;
  const v4u* b4 = (const v4u*)bT;
  const int ml = wm * 16 + l16;

  for (int k0 = 0; k0 < K; k0 += TILE_K) {
    __syncthreads();
    // ---- stage A: 64x32 f16 weight tile ----
    if (useAsync) {
      // one 16B async copy per thread; LDS byte addr = low 32 bits of generic ptr
      unsigned ldsA = (unsigned)(size_t)(void*)aDst;
      const _Float16* g = wrow + k0 + kOffA;
      asm volatile("global_load_async_to_lds_b128 %0, %1, off"
                   :: "v"(ldsA), "v"(g) : "memory");
    } else {
#pragma unroll
      for (int t = 0; t < 8; ++t) {
        int k = k0 + kOffA + t;
        aDst[t] = (mOK && k < K) ? wrow[k] : (_Float16)0.0f;
      }
    }
    // ---- stage B: 128x32 f16 im2col tile (overlaps async A) ----
#pragma unroll
    for (int t = 0; t < 16; ++t) {
      int k = k0 + kOffB + t;
      _Float16 v = (_Float16)0.0f;
      if (nOK && k < K) {
        int c = k / RS; int rs = k - c * RS; int r = rs / S; int s = rs - r * S;
        int ih = ihBase + r, iw = iwBase + s;
        if ((unsigned)ih < (unsigned)IH && (unsigned)iw < (unsigned)IW)
          v = xb[((size_t)c * IH + ih) * IW + iw];
      }
      bDst[t] = v;
    }
    if (useAsync)
      asm volatile("s_wait_asynccnt 0" ::: "memory");
    __syncthreads();

    // ---- fragments (two ds_load_b128 each) + 4 WMMAs ----
    FragH fa, fb;
    fa.q[0] = a4[ml * 4 + ha];
    fa.q[1] = a4[ml * 4 + 2 + ha];
#pragma unroll
    for (int q = 0; q < 4; ++q) {
      int nn = wn * 64 + q * 16 + l16;
      fb.q[0] = b4[nn * 4 + ha];
      fb.q[1] = b4[nn * 4 + 2 + ha];
      acc[q] = __builtin_amdgcn_wmma_f32_16x16x32_f16(false, fa.h, false, fb.h,
                                                      (short)0, acc[q], false, false);
    }
  }

  // ---- epilogue: folded BN / bias, residual add, ReLU, f16 store ----
#pragma unroll
  for (int q = 0; q < 4; ++q) {
    int n = nBase + wn * 64 + q * 16 + l16;
    if (n >= Ntot) continue;
    int eow = n % OW; int t2 = n / OW; int eoh = t2 % OH; int eb = t2 / OH;
#pragma unroll
    for (int r = 0; r < 8; ++r) {
      int ch = mBase + wm * 16 + r + hiM;
      if (ch >= M) continue;
      float v = acc[q][r];
      if (scale) v *= scale[ch];
      if (shift) v += shift[ch];
      size_t o = (((size_t)eb * M + ch) * OH + eoh) * OW + eow;
      if (res) v += (float)res[o];
      if (relu) v = v > 0.f ? v : 0.f;
      out[o] = (_Float16)v;
    }
  }
}

// ---------------- LaneATT assembly kernels (f16 in, f32 out) ----------------
__global__ void lfm_kernel(const _Float16* __restrict__ lf, const _Float16* __restrict__ lfl,
                           const _Float16* __restrict__ lfr, float* __restrict__ out, int total) {
  int i = blockIdx.x * blockDim.x + threadIdx.x;
  if (i >= total) return;
  int w = i % 20; int t = i / 20; int h = t % 12;
  float v;
  if (w == 0 && h < 11)        v = (float)lfl[i];
  else if (w == 19 && h < 11)  v = (float)lfr[i];
  else                          v = (float)lf[i];
  out[i] = v;
}

__global__ void prop_kernel(const _Float16* __restrict__ lf, const _Float16* __restrict__ lfl,
                            const _Float16* __restrict__ lfr, float* __restrict__ out) {
  const int total = 8 * 714 * 77;
  int i = blockIdx.x * blockDim.x + threadIdx.x;
  if (i >= total) return;
  int e = i % 77; int t = i / 77; int p = t % 714; int b = t / 714;
  int a = p / 42; int j = p - a * 42;
  float x0, y0;
  if (j < 11)      { x0 = 0.f; y0 = (float)j / 11.f; }
  else if (j < 22) { x0 = 1.f; y0 = (float)(j - 11) / 11.f; }
  else             { x0 = (float)(j - 22) / 19.f; y0 = 1.f; }
  float val;
  if (e == 2)      val = y0;
  else if (e == 3) val = x0 * 639.f;
  else {
    int c = (e < 2) ? e : (e - 2);
    int ch = a * 75 + c;
    const _Float16* src; int h, w;
    if (j < 11)      { src = lfl; h = j;       w = 0;      }
    else if (j < 22) { src = lfr; h = j - 11;  w = 19;     }
    else             { src = lf;  h = 11;      w = j - 22; }
    float lp = (float)src[(((size_t)b * 1275 + ch) * 12 + h) * 20 + w];
    if (e < 2)       val = lp;
    else if (e == 4) val = lp;  // anchor col 4 is zero
    else {
      int s = e - 5;
      float ys = 1.f - (float)s / 71.f;
      float tanA = tanf(cAngles[a] * PI_F / 180.f);
      val = lp + (x0 + (ys - y0) / tanA) * 639.f;
    }
  }
  out[i] = val;
}

// ---------------- host orchestration ----------------
// Input flattening: x first, then jax tree leaves of params (dict keys sorted):
//  1..4 bn1{beta,gamma,mean,var}; 5 conv1;
//  6..11 conv_bottom_b/w, conv_left_b/w, conv_right_b/w; 12..14 head_{bottom,left,right}_w;
//  15.. stages: per block [bn1 x4][bn2 x4][conv1][conv2]([down_bn x4][down_conv])

extern "C" void kernel_launch(void* const* d_in, const int* in_sizes, int n_in,
                              void* d_out, int out_size, void* d_ws, size_t ws_size,
                              hipStream_t stream) {
  (void)in_sizes; (void)n_in; (void)out_size; (void)ws_size;
  const float* X = (const float*)d_in[0];
  auto F = [&](int i) { return (const float*)d_in[i]; };

  char* wp = (char*)d_ws;
  auto alloc = [&](size_t bytes) -> void* {
    void* r = (void*)wp;
    wp += (bytes + 255) & ~(size_t)255;
    return r;
  };
  auto cvt = [&](const float* w, int n) -> _Float16* {
    _Float16* o = (_Float16*)alloc((size_t)n * sizeof(_Float16));
    cvt_f32_f16_kernel<<<(n + 255) / 256, 256, 0, stream>>>(w, o, n);
    return o;
  };
  auto fold = [&](int i0, int c) -> float* {  // leaves: beta,gamma,mean,var
    float* sc = (float*)alloc((size_t)2 * c * sizeof(float));
    bnfold_kernel<<<(c + 255) / 256, 256, 0, stream>>>(
        F(i0 + 1), F(i0), F(i0 + 2), F(i0 + 3), sc, sc + c, c);
    return sc;
  };
  auto run_conv = [&](const _Float16* x, const _Float16* w, const float* scale,
                      const float* shift, const _Float16* res, _Float16* out,
                      int b, int c, int ih, int iw, int m, int oh, int ow,
                      int r, int strd, int relu) {
    int N = b * oh * ow;
    dim3 g((N + TILE_N - 1) / TILE_N, (m + TILE_M - 1) / TILE_M, 1);
    if (r == 7)
      conv_wmma_kernel<7, 7, 2, 3><<<g, 256, 0, stream>>>(x, w, scale, shift, res, out,
                                                          b, c, ih, iw, m, oh, ow, relu);
    else if (r == 3 && strd == 1)
      conv_wmma_kernel<3, 3, 1, 1><<<g, 256, 0, stream>>>(x, w, scale, shift, res, out,
                                                          b, c, ih, iw, m, oh, ow, relu);
    else if (r == 3)
      conv_wmma_kernel<3, 3, 2, 1><<<g, 256, 0, stream>>>(x, w, scale, shift, res, out,
                                                          b, c, ih, iw, m, oh, ow, relu);
    else if (strd == 1)
      conv_wmma_kernel<1, 1, 1, 0><<<g, 256, 0, stream>>>(x, w, scale, shift, res, out,
                                                          b, c, ih, iw, m, oh, ow, relu);
    else
      conv_wmma_kernel<1, 1, 2, 0><<<g, 256, 0, stream>>>(x, w, scale, shift, res, out,
                                                          b, c, ih, iw, m, oh, ow, relu);
  };

  const int B = 8;
  // f16 activation buffers
  _Float16* x16   = (_Float16*)alloc((size_t)B * 3 * 360 * 640 * sizeof(_Float16));
  _Float16* bufC1 = (_Float16*)alloc((size_t)B * 64 * 180 * 320 * sizeof(_Float16));
  _Float16* buf[4];
  for (int i = 0; i < 4; ++i)
    buf[i] = (_Float16*)alloc((size_t)B * 64 * 90 * 160 * sizeof(_Float16));
  _Float16* lf   = (_Float16*)alloc((size_t)B * 1275 * 240 * sizeof(_Float16));
  _Float16* lflB = (_Float16*)alloc((size_t)B * 1275 * 240 * sizeof(_Float16));
  _Float16* lfrB = (_Float16*)alloc((size_t)B * 1275 * 240 * sizeof(_Float16));
  _Float16* ftmp = (_Float16*)alloc((size_t)B * 64 * 240 * sizeof(_Float16));

  // input f32 -> f16 once
  {
    int n = B * 3 * 360 * 640;
    cvt_f32_f16_kernel<<<(n + 255) / 256, 256, 0, stream>>>(X, x16, n);
  }

  // conv1 + bn + relu, maxpool
  _Float16* wc1 = cvt(F(5), 64 * 3 * 49);
  float* bnTop = fold(1, 64);
  run_conv(x16, wc1, bnTop, bnTop + 64, nullptr, bufC1,
           B, 3, 360, 640, 64, 180, 320, 7, 2, 1);
  {
    int tot = B * 64 * 90 * 160;
    maxpool_kernel<<<(tot + 255) / 256, 256, 0, stream>>>(bufC1, buf[0], B, 64, 180, 320, 90, 160);
  }

  struct BlkDesc { int base, cin, cout, stride; bool down; };
  const BlkDesc blks[8] = {
    {15,  64,  64, 1, false}, {25,  64,  64, 1, false},
    {35,  64, 128, 2, true }, {50, 128, 128, 1, false},
    {60, 128, 256, 2, true }, {75, 256, 256, 1, false},
    {85, 256, 512, 2, true }, {100,512, 512, 1, false},
  };
  _Float16* cur = buf[0];
  int curbuf = 0, H = 90, W = 160;
  for (int bi = 0; bi < 8; ++bi) {
    const BlkDesc& d = blks[bi];
    int OHn = (H - 1) / d.stride + 1;
    int OWn = (W - 1) / d.stride + 1;
    _Float16* w1 = cvt(F(d.base + 8), d.cout * d.cin * 9);
    _Float16* w2 = cvt(F(d.base + 9), d.cout * d.cout * 9);
    float* s1 = fold(d.base, d.cout);
    float* s2 = fold(d.base + 4, d.cout);
    _Float16* t = buf[(curbuf + 1) & 3];
    _Float16* y = buf[(curbuf + 2) & 3];
    const _Float16* resid = cur;
    if (d.down) {
      _Float16* wd = cvt(F(d.base + 14), d.cout * d.cin);
      float* sd = fold(d.base + 10, d.cout);
      _Float16* dbuf = buf[(curbuf + 3) & 3];
      run_conv(cur, wd, sd, sd + d.cout, nullptr, dbuf,
               B, d.cin, H, W, d.cout, OHn, OWn, 1, d.stride, 0);
      resid = dbuf;
    }
    run_conv(cur, w1, s1, s1 + d.cout, nullptr, t,
             B, d.cin, H, W, d.cout, OHn, OWn, 3, d.stride, 1);
    run_conv(t, w2, s2, s2 + d.cout, resid, y,
             B, d.cout, OHn, OWn, d.cout, OHn, OWn, 3, 1, 1);
    cur = y;
    curbuf = (curbuf + 2) & 3;
    H = OHn; W = OWn;
  }

  // heads
  const int headIdx[3][3] = {{6, 7, 12}, {8, 9, 13}, {10, 11, 14}};
  _Float16* headOut[3] = {lf, lflB, lfrB};
  for (int s = 0; s < 3; ++s) {
    _Float16* wcs = cvt(F(headIdx[s][1]), 64 * 512);
    _Float16* whd = cvt(F(headIdx[s][2]), 1275 * 64);
    run_conv(cur, wcs, nullptr, F(headIdx[s][0]), nullptr, ftmp,
             B, 512, 12, 20, 64, 12, 20, 1, 1, 0);
    run_conv(ftmp, whd, nullptr, nullptr, nullptr, headOut[s],
             B, 64, 12, 20, 1275, 12, 20, 1, 1, 0);
  }

  // final assembly: proposals then lfm
  float* outF = (float*)d_out;
  const int propN = 8 * 714 * 77;
  prop_kernel<<<(propN + 255) / 256, 256, 0, stream>>>(lf, lflB, lfrB, outF);
  const int lfmN = 8 * 1275 * 12 * 20;
  lfm_kernel<<<(lfmN + 255) / 256, 256, 0, stream>>>(lf, lflB, lfrB, outF + propN, lfmN);
}
